// _NonLocalBlockND_12781822673578
// MI455X (gfx1250) — matrix operand order repaired and checked by reading, not compiled, not run on hardware
//
#include <hip/hip_runtime.h>

// ---------------- problem constants ----------------
#define B_   8
#define C_   1024
#define CI_  512
#define D_   256          // per-head channels (CI/2)
#define NT_  6272         // 8*28*28 query tokens
#define MT_  1568         // 8*14*14 kv tokens
#define EPS_ 1e-5f
#define SCALE_ 0.0625f    // 256^-0.5

#define PH_STR 264        // LDS row stride (elems) for 32x256 ph tile (pad 8)
#define G_STR  40         // LDS row stride (elems) for 256x32 g tile (pad 8)

typedef __bf16 bf16_t;
typedef __attribute__((ext_vector_type(16))) __bf16 v16bf;
typedef __attribute__((ext_vector_type(8)))  float  v8f;
typedef unsigned int u32x4 __attribute__((ext_vector_type(4)));
typedef unsigned int u32x2 __attribute__((ext_vector_type(2)));
typedef __attribute__((address_space(3))) bf16_t lds_bf16;

__device__ __forceinline__ v8f wmma_bf16(v16bf a, v16bf b, v8f c) {
  // D = A(16x32) * B(32x16) + C, f32 accumulate
  return __builtin_amdgcn_wmma_f32_16x16x32_bf16(false, a, false, b, (short)0, c,
                                                 false, false);
}

// generic (shared) pointer -> 32-bit LDS byte offset
__device__ __forceinline__ unsigned ldsaddr(bf16_t* p) {
  return (unsigned)(unsigned long long)(lds_bf16*)p;
}

// A fragment (16 rows x 32 K, bf16). Source rows K-contiguous, stride in elems.
// lane l: row = l&15, contiguous K chunk [16h .. 16h+15]  (h = l>>4)
__device__ __forceinline__ v16bf ld_frag_a(const bf16_t* base, int stride) {
  const int l = threadIdx.x & 31;
  const bf16_t* p = base + (size_t)(l & 15) * stride + ((l >> 4) << 4);
  union { u32x4 u[2]; v16bf v; } t;
  t.u[0] = *(const u32x4*)(p);
  t.u[1] = *(const u32x4*)(p + 8);
  return t.v;
}

// same but 32-bit address math (for LDS tiles)
__device__ __forceinline__ v16bf ld_frag_a_lds(const bf16_t* base, int stride) {
  const int l = threadIdx.x & 31;
  const bf16_t* p = base + (l & 15) * stride + ((l >> 4) << 4);
  union { u32x4 u[2]; v16bf v; } t;
  t.u[0] = *(const u32x4*)(p);
  t.u[1] = *(const u32x4*)(p + 8);
  return t.v;
}

// B fragment (32 K x 16 cols, bf16). Source cols K-contiguous, stride in elems.
// lane l: col = l&15, contiguous K chunk [16h .. 16h+15]
__device__ __forceinline__ v16bf ld_frag_b(const bf16_t* base, int stride) {
  const int l = threadIdx.x & 31;
  const bf16_t* p = base + (size_t)(l & 15) * stride + ((l >> 4) << 4);
  union { u32x4 u[2]; v16bf v; } t;
  t.u[0] = *(const u32x4*)(p);
  t.u[1] = *(const u32x4*)(p + 8);
  return t.v;
}

// ---------------- small prep kernels ----------------
__global__ void k_pack(const float* __restrict__ s, bf16_t* __restrict__ d, int n) {
  int i = blockIdx.x * 256 + threadIdx.x;
  if (i < n) d[i] = (bf16_t)s[i];
}

__global__ void k_bnprep(const float* __restrict__ gamma, const float* __restrict__ beta,
                         const float* __restrict__ rmean, const float* __restrict__ rvar,
                         float* __restrict__ inv, float* __restrict__ shf) {
  int i = blockIdx.x * 256 + threadIdx.x;
  if (i < C_) {
    float iv = gamma[i] * rsqrtf(rvar[i] + EPS_);
    inv[i] = iv;
    shf[i] = beta[i] - rmean[i] * iv;
  }
}

// x [b][c][p] f32  ->  x_t [b][p][c] bf16   (32x32 LDS tile transpose)
__global__ void __launch_bounds__(256) k_transpose(const float* __restrict__ x,
                                                   bf16_t* __restrict__ xt) {
  __shared__ float tile[32][33];
  const int b = blockIdx.z, c0 = blockIdx.y * 32, p0 = blockIdx.x * 32;
  const int px = threadIdx.x & 31, cy = threadIdx.x >> 5;       // cy: 0..7
  const float* src = x + ((size_t)b * C_ + c0 + cy) * NT_ + p0 + px;
#pragma unroll
  for (int k = 0; k < 4; ++k) tile[cy + 8 * k][px] = src[(size_t)(8 * k) * NT_];
  __syncthreads();
  const int pr = threadIdx.x >> 3, cg = threadIdx.x & 7;        // pr: 0..31, cg: 0..7
  union { bf16_t h4[4]; u32x2 u; } pk;
#pragma unroll
  for (int j = 0; j < 4; ++j) pk.h4[j] = (bf16_t)tile[cg * 4 + j][pr];
  *(u32x2*)(xt + ((size_t)b * NT_ + p0 + pr) * C_ + c0 + cg * 4) = pk.u;
}

// maxpool(1,2,2) on x_t -> xp_t [b][pp][c] bf16 (fully coalesced along c)
__global__ void __launch_bounds__(256) k_pool(const bf16_t* __restrict__ xt,
                                              bf16_t* __restrict__ xpt) {
  const int pp = blockIdx.x, b = blockIdx.y;
  const int tt = pp / 196, r = pp % 196, hh = r / 14, ww = r % 14;
  const int p = tt * 784 + hh * 2 * 28 + ww * 2;
  const bf16_t* r0 = xt + ((size_t)b * NT_ + p) * C_;
  const bf16_t* r1 = r0 + 28 * C_;
  bf16_t* dst = xpt + ((size_t)b * MT_ + pp) * C_;
#pragma unroll
  for (int c = threadIdx.x; c < C_; c += 256) {
    float a = fmaxf(fmaxf((float)r0[c], (float)r0[c + C_]),
                    fmaxf((float)r1[c], (float)r1[c + C_]));
    dst[c] = (bf16_t)a;
  }
}

// ---------------- conv-as-GEMM (WMMA), 16o x 32p per wave ----------------
// Y[o,p] = sum_c W[o,c] * X[b][p][c] + bias[o]
// out_tr=1: store transposed Y[b][p][o] (b128); out_tr=0: natural Y[b][o][p] (b16)
__global__ void __launch_bounds__(128) k_conv(const bf16_t* __restrict__ W,
                                              const float* __restrict__ bias,
                                              const bf16_t* __restrict__ X,
                                              bf16_t* __restrict__ Y,
                                              int M, int N, int K, int out_tr) {
  const int b = blockIdx.z;
  const int o0 = blockIdx.y * 64 + (threadIdx.x >> 5) * 16;
  const int p0 = blockIdx.x * 32;
  const bf16_t* Wr = W + (size_t)o0 * K;
  const bf16_t* Xr = X + ((size_t)b * N + p0) * K;
  v8f acc0 = {}, acc1 = {};
#pragma unroll 2
  for (int k0 = 0; k0 < K; k0 += 32) {
    v16bf a = ld_frag_a(Wr + k0, K);
    acc0 = wmma_bf16(a, ld_frag_b(Xr + k0, K), acc0);
    acc1 = wmma_bf16(a, ld_frag_b(Xr + (size_t)16 * K + k0, K), acc1);
  }
  const int l = threadIdx.x & 31, q = l & 15, hf = l >> 4;
  const int ob = o0 + 8 * hf;
  v8f accs[2] = {acc0, acc1};
#pragma unroll
  for (int s = 0; s < 2; ++s) {
    const int p = p0 + s * 16 + q;
    if (out_tr) {
      union { bf16_t h8[8]; u32x4 u; } pk;
#pragma unroll
      for (int v = 0; v < 8; ++v) pk.h8[v] = (bf16_t)(accs[s][v] + bias[ob + v]);
      *(u32x4*)(Y + ((size_t)b * N + p) * M + ob) = pk.u;
    } else {
#pragma unroll
      for (int v = 0; v < 8; ++v)
        Y[((size_t)b * M + ob + v) * N + p] = (bf16_t)(accs[s][v] + bias[ob + v]);
    }
  }
}

// ---------------- fused flash attention ----------------
// 4 waves/block share one head; ph/g kv-tiles staged in LDS via async DMA loads.
// scores computed transposed: f^T = ph^T(m x d) * th(d x n) -> lane owns one query col
// yacc = y^T (256 d-rows x 16 n-cols per wave) accumulated over kv tiles of 32
__global__ void __launch_bounds__(128) k_attn(const bf16_t* __restrict__ th_t,
                                              const bf16_t* __restrict__ ph_t,
                                              const bf16_t* __restrict__ g_n,
                                              bf16_t* __restrict__ y_t) {
  alignas(16) __shared__ bf16_t s_ph[32 * PH_STR];   // 16.9 KB: 32 kv rows x 256 d
  alignas(16) __shared__ bf16_t s_g[256 * G_STR];    // 20.5 KB: 256 d rows x 32 kv

  const int bh = blockIdx.y;             // 0..15
  const int b = bh >> 1, hh = bh & 1;
  const int tid = threadIdx.x;
  const int wv = tid >> 5;
  const int nt = blockIdx.x * 4 + wv;    // this wave's 16-query tile
  const int l = tid & 31, q = l & 15, hf = l >> 4;

  const bf16_t* TH = th_t + ((size_t)b * NT_ + (size_t)nt * 16) * CI_ + hh * D_;
  const bf16_t* PH = ph_t + (size_t)b * MT_ * CI_ + hh * D_;
  const bf16_t* G  = g_n  + ((size_t)b * CI_ + (size_t)hh * D_) * MT_;

  // preload query B-fragments (16 cols x K=256) -> 8 frags, kept in VGPRs
  v16bf thB[8];
#pragma unroll
  for (int ks = 0; ks < 8; ++ks) thB[ks] = ld_frag_b(TH + ks * 32, CI_);

  v8f yacc[16];
#pragma unroll
  for (int t = 0; t < 16; ++t) yacc[t] = (v8f){};

  float mx = -3.0e38f, sm = 0.0f;

  for (int m0 = 0; m0 < MT_; m0 += 32) {
    // ---- async DMA stage of ph (32x256) and g (256x32) kv tiles into LDS ----
    // (ASYNCcnt path: global_load_async_to_lds_b128, no VGPR round trip)
#pragma unroll
    for (int j = 0; j < 8; ++j) {
      const int idx = tid + 128 * j;            // 1024 16B-chunks
      const int row = idx >> 5, col = idx & 31;
      const unsigned dst = ldsaddr(&s_ph[row * PH_STR + col * 8]);
      const bf16_t* src = PH + (size_t)(m0 + row) * CI_ + col * 8;
      asm volatile("global_load_async_to_lds_b128 %0, %1, off"
                   :: "v"(dst), "v"(src) : "memory");
    }
#pragma unroll
    for (int j = 0; j < 8; ++j) {
      const int idx = tid + 128 * j;            // 1024 16B-chunks
      const int row = idx >> 2, c4 = idx & 3;
      const unsigned dst = ldsaddr(&s_g[row * G_STR + c4 * 8]);
      const bf16_t* src = G + (size_t)row * MT_ + m0 + c4 * 8;
      asm volatile("global_load_async_to_lds_b128 %0, %1, off"
                   :: "v"(dst), "v"(src) : "memory");
    }
    asm volatile("s_wait_asynccnt 0x0" ::: "memory");  // own transfers done
    __syncthreads();                                   // all waves' transfers done

    // ---- scores f^T tile (32 m x 16 n) as two 16x16 C-frags ----
    v8f c0 = {}, c1 = {};
#pragma unroll
    for (int ks = 0; ks < 8; ++ks)
      c0 = wmma_bf16(ld_frag_a_lds(s_ph + ks * 32, PH_STR), thB[ks], c0);
#pragma unroll
    for (int ks = 0; ks < 8; ++ks)
      c1 = wmma_bf16(ld_frag_a_lds(s_ph + 16 * PH_STR + ks * 32, PH_STR), thB[ks], c1);

    float p[16];
#pragma unroll
    for (int v = 0; v < 8; ++v) { p[v] = c0[v] * SCALE_; p[8 + v] = c1[v] * SCALE_; }

    // ---- online softmax (per query column) ----
    float lm = p[0];
#pragma unroll
    for (int k = 1; k < 16; ++k) lm = fmaxf(lm, p[k]);
    lm = fmaxf(lm, __shfl_xor(lm, 16, 32));
    const float nmx = fmaxf(mx, lm);
    const float corr = __expf(mx - nmx);
    mx = nmx;
    float ls = 0.0f;
#pragma unroll
    for (int k = 0; k < 16; ++k) { p[k] = __expf(p[k] - mx); ls += p[k]; }
    ls += __shfl_xor(ls, 16, 32);
    sm = sm * corr + ls;
#pragma unroll
    for (int t = 0; t < 16; ++t)
#pragma unroll
      for (int v = 0; v < 8; ++v) yacc[t][v] *= corr;

    // ---- build P^T B-fragment (K=32 kv, 16 cols) from score C-frags ----
    // lane half hf covers K = 16*hf .. 16*hf+15; cross-half values via shfl_xor(16)
    union { bf16_t h16[16]; v16bf v; } pbu;
#pragma unroll
    for (int k = 0; k < 8; ++k) {
      const float oth_lo = __shfl_xor(p[k], 16, 32);
      const float oth_hi = __shfl_xor(p[8 + k], 16, 32);
      const float s_lo = hf ? oth_hi : p[k];              // hf=0: mu=k   | hf=1: mu=16+k
      const float s_hi = hf ? p[8 + k] : oth_lo;          // hf=0: mu=8+k | hf=1: mu=24+k
      pbu.h16[k]     = (bf16_t)s_lo;
      pbu.h16[8 + k] = (bf16_t)s_hi;
    }

    // ---- y^T += g(d x m_tile) * P^T(m_tile x n) ----
#pragma unroll
    for (int t = 0; t < 16; ++t)
      yacc[t] = wmma_bf16(ld_frag_a_lds(s_g + (t * 16) * G_STR, G_STR), pbu.v, yacc[t]);

    __syncthreads();   // protect LDS tiles before next overwrite
  }

  // ---- normalize, pack bf16, store y^T -> y_t[b][n][ci] ----
  const float isv = 1.0f / sm;
  bf16_t* yb = y_t + ((size_t)b * NT_ + (size_t)nt * 16 + q) * CI_ + hh * D_ + 8 * hf;
#pragma unroll
  for (int t = 0; t < 16; ++t) {
    union { bf16_t h8[8]; u32x4 u; } pk;
#pragma unroll
    for (int v = 0; v < 8; ++v) pk.h8[v] = (bf16_t)(yacc[t][v] * isv);
    *(u32x4*)(yb + t * 16) = pk.u;
  }
}

// ---------------- final conv + BN + residual (f32 out), 16o x 32p per wave ----------------
__global__ void __launch_bounds__(128) k_outconv(const bf16_t* __restrict__ W,
                                                 const float* __restrict__ bw,
                                                 const bf16_t* __restrict__ Yt,
                                                 const float* __restrict__ inv,
                                                 const float* __restrict__ shf,
                                                 const float* __restrict__ x,
                                                 float* __restrict__ out) {
  const int b = blockIdx.z;
  const int o0 = blockIdx.y * 64 + (threadIdx.x >> 5) * 16;
  const int p0 = blockIdx.x * 32;
  const bf16_t* Wr = W + (size_t)o0 * CI_;
  const bf16_t* Yr = Yt + ((size_t)b * NT_ + p0) * CI_;
  v8f acc0 = {}, acc1 = {};
#pragma unroll 2
  for (int k0 = 0; k0 < CI_; k0 += 32) {
    v16bf a = ld_frag_a(Wr + k0, CI_);
    acc0 = wmma_bf16(a, ld_frag_b(Yr + k0, CI_), acc0);
    acc1 = wmma_bf16(a, ld_frag_b(Yr + (size_t)16 * CI_ + k0, CI_), acc1);
  }
  const int l = threadIdx.x & 31, q = l & 15, hf = l >> 4;
  v8f accs[2] = {acc0, acc1};
#pragma unroll
  for (int s = 0; s < 2; ++s) {
    const int p = p0 + s * 16 + q;
#pragma unroll
    for (int v = 0; v < 8; ++v) {
      const int o = o0 + 8 * hf + v;
      const size_t idx = ((size_t)b * C_ + o) * NT_ + p;
      out[idx] = (accs[s][v] + bw[o]) * inv[o] + shf[o] + x[idx];
    }
  }
}

// ---------------- host launcher ----------------
extern "C" void kernel_launch(void* const* d_in, const int* in_sizes, int n_in,
                              void* d_out, int out_size, void* d_ws, size_t ws_size,
                              hipStream_t stream) {
  (void)in_sizes; (void)n_in; (void)out_size; (void)ws_size;
  const float* x     = (const float*)d_in[0];
  const float* Wg    = (const float*)d_in[1];
  const float* bg    = (const float*)d_in[2];
  const float* Wth   = (const float*)d_in[3];
  const float* bth   = (const float*)d_in[4];
  const float* Wph   = (const float*)d_in[5];
  const float* bph   = (const float*)d_in[6];
  const float* Ww    = (const float*)d_in[7];
  const float* bw    = (const float*)d_in[8];
  const float* gamma = (const float*)d_in[9];
  const float* beta  = (const float*)d_in[10];
  const float* rmean = (const float*)d_in[11];
  const float* rvar  = (const float*)d_in[12];
  float* out = (float*)d_out;

  char* ws = (char*)d_ws;
  size_t off = 0;
  const size_t WEL = (size_t)CI_ * C_;             // 524288
  bf16_t* Wg_b  = (bf16_t*)(ws + off); off += WEL * 2;
  bf16_t* Wth_b = (bf16_t*)(ws + off); off += WEL * 2;
  bf16_t* Wph_b = (bf16_t*)(ws + off); off += WEL * 2;
  bf16_t* Ww_b  = (bf16_t*)(ws + off); off += WEL * 2;
  float*  inv   = (float*)(ws + off);  off += C_ * 4;
  float*  shf   = (float*)(ws + off);  off += C_ * 4;
  bf16_t* x_t   = (bf16_t*)(ws + off); off += (size_t)B_ * NT_ * C_ * 2;   // 102.8 MB
  bf16_t* xp_t  = (bf16_t*)(ws + off); off += (size_t)B_ * MT_ * C_ * 2;   // 25.7 MB
  bf16_t* th_t  = (bf16_t*)(ws + off); off += (size_t)B_ * NT_ * CI_ * 2;  // 51.4 MB
  bf16_t* ph_t  = (bf16_t*)(ws + off); off += (size_t)B_ * MT_ * CI_ * 2;  // 12.9 MB
  bf16_t* g_n   = (bf16_t*)(ws + off); off += (size_t)B_ * MT_ * CI_ * 2;  // 12.9 MB
  bf16_t* y_t   = x_t;  // x_t dead after th-conv; alias to save workspace

  // weights -> bf16, BN fold
  k_pack<<<dim3((unsigned)((WEL + 255) / 256)), 256, 0, stream>>>(Wg,  Wg_b,  (int)WEL);
  k_pack<<<dim3((unsigned)((WEL + 255) / 256)), 256, 0, stream>>>(Wth, Wth_b, (int)WEL);
  k_pack<<<dim3((unsigned)((WEL + 255) / 256)), 256, 0, stream>>>(Wph, Wph_b, (int)WEL);
  k_pack<<<dim3((unsigned)((WEL + 255) / 256)), 256, 0, stream>>>(Ww,  Ww_b,  (int)WEL);
  k_bnprep<<<dim3(4), 256, 0, stream>>>(gamma, beta, rmean, rvar, inv, shf);

  // activation layout: transpose + pool
  k_transpose<<<dim3(NT_ / 32, C_ / 32, B_), 256, 0, stream>>>(x, x_t);
  k_pool<<<dim3(MT_, B_), 256, 0, stream>>>(x_t, xp_t);

  // projections (WMMA GEMMs)
  k_conv<<<dim3(NT_ / 32, CI_ / 64, B_), 128, 0, stream>>>(Wth_b, bth, x_t,  th_t, CI_, NT_, C_, 1);
  k_conv<<<dim3(MT_ / 32, CI_ / 64, B_), 128, 0, stream>>>(Wph_b, bph, xp_t, ph_t, CI_, MT_, C_, 1);
  k_conv<<<dim3(MT_ / 32, CI_ / 64, B_), 128, 0, stream>>>(Wg_b,  bg,  xp_t, g_n,  CI_, MT_, C_, 0);

  // fused flash attention (scores + softmax + PV), 4 query-tiles per block
  k_attn<<<dim3(NT_ / 64, B_ * 2), 128, 0, stream>>>(th_t, ph_t, g_n, y_t);

  // output projection + BN + residual
  k_outconv<<<dim3(NT_ / 32, C_ / 64, B_), 128, 0, stream>>>(Ww_b, bw, y_t, inv, shf, x, out);
}
